// DirectedCellGraphTransformer_86655260164297
// MI455X (gfx1250) — matrix operand order repaired and checked by reading, compiled, not run on hardware
//
#include <hip/hip_runtime.h>
#include <hip/hip_bf16.h>

// ---------------------------------------------------------------------------
// RGAT GNN forward for MI455X (gfx1250, wave32, WMMA).
//   N=20480 nodes, E=163840 edges, G=512 graphs, D=128, H=4, R=8, L=4, HK=512
// Strategy:
//   * bf16 WMMA GEMM for x_all = x @ W[r]  (the 86 GFLOP hot loop)
//   * x_all stored bf16 -> halves HBM traffic of GEMM-write + edge gather
//   * q/k via fused WQ[r] = W[r]@Q (exact algebra) -> no re-read of x_all
//   * edge softmax with float atomic max/add, bf16 message scatter-add (f32 atomics)
// ---------------------------------------------------------------------------

constexpr int NN  = 20480;
constexpr int EE  = 163840;
constexpr int GG  = 512;
constexpr int DD  = 128;
constexpr int HH  = 4;
constexpr int RR  = 8;
constexpr int LL  = 4;
constexpr int HKK = 512;          // H * D
constexpr int NPG = NN / GG;      // 40 nodes per graph (contiguous)
constexpr float NEG_SLOPE = 0.2f;

typedef __attribute__((ext_vector_type(16))) __bf16 v16bf;
typedef __attribute__((ext_vector_type(8)))  __bf16 v8bf;
typedef __attribute__((ext_vector_type(8)))  float  v8f;

union AFrag { v16bf v; v8bf h[2]; };

// fp32 -> bf16 (round to nearest even), bf16 -> fp32
__device__ __forceinline__ unsigned short f2bf(float f) {
    unsigned u = __float_as_uint(f);
    u += 0x7FFFu + ((u >> 16) & 1u);
    return (unsigned short)(u >> 16);
}
__device__ __forceinline__ float bf2f(unsigned short h) {
    return __uint_as_float(((unsigned)h) << 16);
}

__device__ __forceinline__ void atomicMaxF(float* addr, float v) {
    if (v >= 0.0f) atomicMax((int*)addr, __float_as_int(v));
    else           atomicMin((unsigned int*)addr, __float_as_uint(v));
}

// ---------------------------------------------------------------------------
// One-time: transpose W [L][R][D][HK] f32 -> Wt [L][R][HK][D] bf16
// ---------------------------------------------------------------------------
__global__ __launch_bounds__(256) void transpose_w_kernel(
    const float* __restrict__ W, unsigned short* __restrict__ Wt)
{
    int i = blockIdx.x * 256 + threadIdx.x;           // over L*R*HK*D
    if (i >= LL * RR * HKK * DD) return;
    int d  = i & (DD - 1);
    int hk = (i >> 7) & (HKK - 1);
    int rl = i >> 16;                                  // l*R + r
    Wt[i] = f2bf(W[((size_t)rl * DD + d) * HKK + hk]); // contiguous write
}

// ---------------------------------------------------------------------------
// Embedding gather: x[n,d] = emb[x_nodes[n], d]  (fp32 + bf16 copies)
// ---------------------------------------------------------------------------
__global__ __launch_bounds__(256) void embed_kernel(
    const int* __restrict__ x_nodes, const float* __restrict__ emb,
    float* __restrict__ x_f, unsigned short* __restrict__ x_bf)
{
    int i = blockIdx.x * 256 + threadIdx.x;           // over N*D
    if (i >= NN * DD) return;
    int n = i >> 7, d = i & (DD - 1);
    float v = emb[x_nodes[n] * DD + d];
    x_f[i]  = v;
    x_bf[i] = f2bf(v);
}

// ---------------------------------------------------------------------------
// Per-layer fused attention projections: WQ[r] = W[l,r] @ Q[l]  (128x4 each)
// ---------------------------------------------------------------------------
__global__ __launch_bounds__(256) void fuse_qk_kernel(
    const float* __restrict__ Wl,   // [R][D][HK] for layer l
    const float* __restrict__ Ql,   // [HK][H]
    const float* __restrict__ Kl,   // [HK][H]
    float* __restrict__ WQ, float* __restrict__ WK)   // [R][D][H]
{
    int i = blockIdx.x * 256 + threadIdx.x;           // over R*D*H = 4096
    if (i >= RR * DD * HH) return;
    int h = i & 3, d = (i >> 2) & (DD - 1), r = i >> 9;
    const float* wrow = Wl + ((size_t)r * DD + d) * HKK;
    float q = 0.0f, k = 0.0f;
    for (int c = 0; c < HKK; ++c) {
        float w = wrow[c];
        q = fmaf(w, Ql[c * HH + h], q);
        k = fmaf(w, Kl[c * HH + h], k);
    }
    WQ[i] = q;
    WK[i] = k;
}

// ---------------------------------------------------------------------------
// Hot GEMM: x_all[n,r,:] = x[n,:] @ W[l,r,:,:]  via bf16 WMMA.
// One wave: 16 rows x 64 cols (4 accumulators, 16 v_wmma per wave).
// Waves: rowTiles(1280) * R(8) * colGroups(8) = 81920 -> 10240 blocks of 256.
// ---------------------------------------------------------------------------
__global__ __launch_bounds__(256) void gemm_xall_kernel(
    const unsigned short* __restrict__ x_bf,   // [N][D] bf16
    const unsigned short* __restrict__ Wt,     // [R][HK][D] bf16 (this layer)
    unsigned short* __restrict__ x_all)        // [N][R][HK] bf16
{
    int w    = (blockIdx.x * 256 + threadIdx.x) >> 5;
    int lane = threadIdx.x & 31;
    int cg   = w & 7;            // 64-col group
    int r    = (w >> 3) & 7;     // relation
    int rt   = w >> 6;           // 16-row tile, 0..1279
    int half = lane >> 4;
    int l15  = lane & 15;

    // A fragment rows: lane l15 holds row (rt*16 + l15); K split by lane half.
    const unsigned short* Ab = x_bf + (size_t)(rt * 16 + l15) * DD;
    // B fragment cols: lane l15 holds column (cg*64 + t*16 + l15); K contiguous
    // along D thanks to the [HK][D] transpose.
    const unsigned short* Bb = Wt + ((size_t)r * HKK + cg * 64 + l15) * DD;

    v8f acc0 = {}, acc1 = {}, acc2 = {}, acc3 = {};

    for (int ks = 0; ks < 4; ++ks) {
        int k0 = ks * 32;
        AFrag a;
        a.h[0] = *(const v8bf*)(Ab + k0 + half * 16);       // K = k0 + half*16 .. +7
        a.h[1] = *(const v8bf*)(Ab + k0 + half * 16 + 8);   // K = k0+half*16+8 ..
        const unsigned short* bp = Bb + k0 + half * 16;     // K = k0 + half*16 .. +15
        AFrag b0, b1, b2, b3;
        b0.h[0] = *(const v8bf*)(bp);               b0.h[1] = *(const v8bf*)(bp + 8);
        b1.h[0] = *(const v8bf*)(bp + 16 * DD);     b1.h[1] = *(const v8bf*)(bp + 16 * DD + 8);
        b2.h[0] = *(const v8bf*)(bp + 32 * DD);     b2.h[1] = *(const v8bf*)(bp + 32 * DD + 8);
        b3.h[0] = *(const v8bf*)(bp + 48 * DD);     b3.h[1] = *(const v8bf*)(bp + 48 * DD + 8);

        acc0 = __builtin_amdgcn_wmma_f32_16x16x32_bf16(false, a.v, false, b0.v,
                                                       (short)0, acc0, false, false);
        acc1 = __builtin_amdgcn_wmma_f32_16x16x32_bf16(false, a.v, false, b1.v,
                                                       (short)0, acc1, false, false);
        acc2 = __builtin_amdgcn_wmma_f32_16x16x32_bf16(false, a.v, false, b2.v,
                                                       (short)0, acc2, false, false);
        acc3 = __builtin_amdgcn_wmma_f32_16x16x32_bf16(false, a.v, false, b3.v,
                                                       (short)0, acc3, false, false);
    }

    // C/D layout: lane holds column (l15), rows (half*8 + v) in VGPR v.
    int colBase = cg * 64 + l15;
    int rowBase = rt * 16 + half * 8;
    v8f accs[4] = { acc0, acc1, acc2, acc3 };
    for (int t = 0; t < 4; ++t) {
        size_t base = ((size_t)rowBase * RR + r) * HKK + colBase + t * 16;
        #pragma unroll
        for (int v = 0; v < 8; ++v)
            x_all[base + (size_t)v * RR * HKK] = f2bf(accs[t][v]);
    }
}

// ---------------------------------------------------------------------------
// q_all/k_all [N][R][H] = x @ WQ[r] / x @ WK[r]   (tiny fused projections)
// ---------------------------------------------------------------------------
__global__ __launch_bounds__(256) void qk_kernel(
    const unsigned short* __restrict__ x_bf,
    const float* __restrict__ WQ, const float* __restrict__ WK,
    float* __restrict__ q_all, float* __restrict__ k_all)
{
    int i = blockIdx.x * 256 + threadIdx.x;    // over N*R*H
    if (i >= NN * RR * HH) return;
    int h = i & 3, r = (i >> 2) & 7, n = i >> 5;
    const unsigned short* xp = x_bf + (size_t)n * DD;
    const float* wq = WQ + ((size_t)r * DD) * HH + h;
    const float* wk = WK + ((size_t)r * DD) * HH + h;
    float q = 0.0f, k = 0.0f;
    for (int d = 0; d < DD; ++d) {
        float xv = bf2f(xp[d]);
        q = fmaf(xv, wq[d * HH], q);
        k = fmaf(xv, wk[d * HH], k);
    }
    q_all[i] = q;
    k_all[i] = k;
}

// ---------------------------------------------------------------------------
// Per-layer reset: agg=0, amax=-inf, denom=0
// ---------------------------------------------------------------------------
__global__ __launch_bounds__(256) void reset_kernel(
    float* __restrict__ agg, float* __restrict__ amax, float* __restrict__ denom)
{
    int i = blockIdx.x * 256 + threadIdx.x;
    if (i < NN * HKK) agg[i] = 0.0f;
    if (i < NN * HH) {
        amax[i]  = __int_as_float(0xFF800000);  // -inf
        denom[i] = 0.0f;
    }
}

// ---------------------------------------------------------------------------
// Edge pass 1: alpha = leaky_relu(q[dst,et] + k[src,et]); segment max over dst
// ---------------------------------------------------------------------------
__global__ __launch_bounds__(256) void edge_alpha_kernel(
    const int* __restrict__ src, const int* __restrict__ dst,
    const int* __restrict__ etype,
    const float* __restrict__ q_all, const float* __restrict__ k_all,
    float* __restrict__ alpha, float* __restrict__ amax)
{
    int e = blockIdx.x * 256 + threadIdx.x;
    if (e >= EE) return;
    int s = src[e], d = dst[e], et = etype[e];
    const float* qv = q_all + ((size_t)d * RR + et) * HH;
    const float* kv = k_all + ((size_t)s * RR + et) * HH;
    #pragma unroll
    for (int h = 0; h < HH; ++h) {
        float a = qv[h] + kv[h];
        a = (a > 0.0f) ? a : NEG_SLOPE * a;
        alpha[(size_t)e * HH + h] = a;
        atomicMaxF(&amax[d * HH + h], a);
    }
}

// ---------------------------------------------------------------------------
// Edge pass 2: ex = exp(alpha - amax[dst]); segment sum over dst
// ---------------------------------------------------------------------------
__global__ __launch_bounds__(256) void edge_exp_kernel(
    const int* __restrict__ dst, const float* __restrict__ amax,
    float* __restrict__ alpha, float* __restrict__ denom)
{
    int e = blockIdx.x * 256 + threadIdx.x;
    if (e >= EE) return;
    int d = dst[e];
    #pragma unroll
    for (int h = 0; h < HH; ++h) {
        float ex = __expf(alpha[(size_t)e * HH + h] - amax[d * HH + h]);
        alpha[(size_t)e * HH + h] = ex;
        atomicAdd(&denom[d * HH + h], ex);
    }
}

// ---------------------------------------------------------------------------
// Edge pass 3: agg[dst] += x_all[src,et] * a    (one block of 128 per edge,
// 4 contiguous channels per thread; bf16 gather, fp32 atomic scatter)
// ---------------------------------------------------------------------------
__global__ __launch_bounds__(128) void edge_msg_kernel(
    const int* __restrict__ src, const int* __restrict__ dst,
    const int* __restrict__ etype,
    const float* __restrict__ alpha, const float* __restrict__ denom,
    const unsigned short* __restrict__ x_all, float* __restrict__ agg)
{
    int e = blockIdx.x;
    int t = threadIdx.x;                       // 0..127 -> channels t*4..t*4+3
    int s = src[e], d = dst[e], et = etype[e];
    int h = t >> 5;                            // (t*4)/128
    float a = alpha[(size_t)e * HH + h] / (denom[d * HH + h] + 1e-16f);

    const unsigned short* xp = x_all + ((size_t)s * RR + et) * HKK + t * 4;
    __builtin_prefetch(xp, 0, 0);
    float* ap = agg + (size_t)d * HKK + t * 4;
    uint2 packed = *(const uint2*)xp;          // 4 bf16
    atomicAdd(&ap[0], bf2f((unsigned short)(packed.x & 0xFFFF))        * a);
    atomicAdd(&ap[1], bf2f((unsigned short)(packed.x >> 16))           * a);
    atomicAdd(&ap[2], bf2f((unsigned short)(packed.y & 0xFFFF))        * a);
    atomicAdd(&ap[3], bf2f((unsigned short)(packed.y >> 16))           * a);
}

// ---------------------------------------------------------------------------
// Layer finalize: x = relu(mean_h(agg) + bias)
// ---------------------------------------------------------------------------
__global__ __launch_bounds__(256) void layer_out_kernel(
    const float* __restrict__ agg, const float* __restrict__ bias,
    float* __restrict__ x_f, unsigned short* __restrict__ x_bf)
{
    int i = blockIdx.x * 256 + threadIdx.x;    // over N*D
    if (i >= NN * DD) return;
    int n = i >> 7, d = i & (DD - 1);
    const float* ag = agg + (size_t)n * HKK + d;
    float v = 0.25f * (ag[0] + ag[DD] + ag[2 * DD] + ag[3 * DD]) + bias[d];
    v = fmaxf(v, 0.0f);
    x_f[i]  = v;
    x_bf[i] = f2bf(v);
}

// ---------------------------------------------------------------------------
// Heads: mean-pool (40 contiguous nodes/graph) -> fc1+relu -> policy / value
// ---------------------------------------------------------------------------
__global__ __launch_bounds__(64) void head_kernel(
    const float* __restrict__ x_f,
    const float* __restrict__ fc1_w, const float* __restrict__ fc1_b,
    const float* __restrict__ pol_w, const float* __restrict__ pol_b,
    const float* __restrict__ val_w, const float* __restrict__ val_b,
    float* __restrict__ out)
{
    __shared__ float pooled[DD];
    __shared__ float hv[64];
    int g = blockIdx.x, t = threadIdx.x;

    for (int d = t; d < DD; d += 64) {
        float s = 0.0f;
        const float* xp = x_f + (size_t)g * NPG * DD + d;
        for (int i = 0; i < NPG; ++i) s += xp[i * DD];
        pooled[d] = s * (1.0f / NPG);
    }
    __syncthreads();

    float s = fc1_b[t];
    for (int d = 0; d < DD; ++d) s = fmaf(pooled[d], fc1_w[d * 64 + t], s);
    hv[t] = fmaxf(s, 0.0f);
    __syncthreads();

    if (t < 7) {
        float p = pol_b[t];
        for (int i = 0; i < 64; ++i) p = fmaf(hv[i], pol_w[i * 7 + t], p);
        out[g * 7 + t] = p;
    } else if (t == 7) {
        float v = val_b[0];
        for (int i = 0; i < 64; ++i) v = fmaf(hv[i], val_w[i], v);
        out[GG * 7 + g] = tanhf(v);
    }
}

// ---------------------------------------------------------------------------
// Launch
// ---------------------------------------------------------------------------
extern "C" void kernel_launch(void* const* d_in, const int* in_sizes, int n_in,
                              void* d_out, int out_size, void* d_ws, size_t ws_size,
                              hipStream_t stream)
{
    const int*   x_nodes = (const int*)  d_in[0];
    const int*   eidx    = (const int*)  d_in[1];   // [2][E]
    const int*   etype   = (const int*)  d_in[2];
    // d_in[3] = batch (contiguous by construction, unused)
    const float* emb     = (const float*)d_in[4];
    const float* Wf      = (const float*)d_in[5];   // [L][R][D][HK]
    const float* Qf      = (const float*)d_in[6];   // [L][HK][H]
    const float* Kf      = (const float*)d_in[7];   // [L][HK][H]
    const float* Bf      = (const float*)d_in[8];   // [L][D]
    const float* fc1_w   = (const float*)d_in[9];
    const float* fc1_b   = (const float*)d_in[10];
    const float* pol_w   = (const float*)d_in[11];
    const float* pol_b   = (const float*)d_in[12];
    const float* val_w   = (const float*)d_in[13];
    const float* val_b   = (const float*)d_in[14];
    float* out = (float*)d_out;

    const int* src = eidx;
    const int* dst = eidx + EE;

    // Workspace carve-up (256B aligned)
    char*  ws = (char*)d_ws;
    size_t o  = 0;
    auto alloc = [&](size_t bytes) -> char* {
        char* p = ws + o;
        o += (bytes + 255) & ~(size_t)255;
        return p;
    };
    float*          x_f   = (float*)         alloc((size_t)NN * DD * 4);
    unsigned short* x_bf  = (unsigned short*)alloc((size_t)NN * DD * 2);
    unsigned short* Wt    = (unsigned short*)alloc((size_t)LL * RR * HKK * DD * 2);
    float*          WQ    = (float*)         alloc((size_t)RR * DD * HH * 4);
    float*          WK    = (float*)         alloc((size_t)RR * DD * HH * 4);
    unsigned short* x_all = (unsigned short*)alloc((size_t)NN * RR * HKK * 2);
    float*          q_all = (float*)         alloc((size_t)NN * RR * HH * 4);
    float*          k_all = (float*)         alloc((size_t)NN * RR * HH * 4);
    float*          alpha = (float*)         alloc((size_t)EE * HH * 4);
    float*          amax  = (float*)         alloc((size_t)NN * HH * 4);
    float*          denom = (float*)         alloc((size_t)NN * HH * 4);
    float*          agg   = (float*)         alloc((size_t)NN * HKK * 4);
    (void)ws_size; (void)n_in; (void)in_sizes; (void)out_size;

    // One-time prep
    transpose_w_kernel<<<(LL * RR * HKK * DD) / 256, 256, 0, stream>>>(Wf, Wt);
    embed_kernel<<<(NN * DD) / 256, 256, 0, stream>>>(x_nodes, emb, x_f, x_bf);

    for (int l = 0; l < LL; ++l) {
        const float*          Wl  = Wf + (size_t)l * RR * DD * HKK;
        const unsigned short* Wtl = Wt + (size_t)l * RR * HKK * DD;

        fuse_qk_kernel<<<(RR * DD * HH) / 256, 256, 0, stream>>>(
            Wl, Qf + (size_t)l * HKK * HH, Kf + (size_t)l * HKK * HH, WQ, WK);

        // 81920 waves = 10240 blocks of 8 waves
        gemm_xall_kernel<<<10240, 256, 0, stream>>>(x_bf, Wtl, x_all);

        qk_kernel<<<(NN * RR * HH) / 256, 256, 0, stream>>>(x_bf, WQ, WK, q_all, k_all);

        reset_kernel<<<(NN * HKK) / 256, 256, 0, stream>>>(agg, amax, denom);

        edge_alpha_kernel<<<EE / 256, 256, 0, stream>>>(src, dst, etype,
                                                        q_all, k_all, alpha, amax);
        edge_exp_kernel<<<EE / 256, 256, 0, stream>>>(dst, amax, alpha, denom);
        edge_msg_kernel<<<EE, 128, 0, stream>>>(src, dst, etype, alpha, denom,
                                                x_all, agg);

        layer_out_kernel<<<(NN * DD) / 256, 256, 0, stream>>>(
            agg, Bf + (size_t)l * DD, x_f, x_bf);
    }

    head_kernel<<<GG, 64, 0, stream>>>(x_f, fc1_w, fc1_b, pol_w, pol_b,
                                       val_w, val_b, out);
}
